// AGNN_52613349376553
// MI455X (gfx1250) — matrix operand order repaired and checked by reading, not compiled, not run on hardware
//
#include <hip/hip_runtime.h>

typedef __attribute__((ext_vector_type(16))) _Float16 v16h;
typedef __attribute__((ext_vector_type(2)))  _Float16 h2v;
typedef __attribute__((ext_vector_type(8)))  float    v8f;

#define EPS_N 1e-12f
#define TPB_TILES 4   // tiles (waves) per 128-thread block in the WMMA kernels

// ---------------------------------------------------------------------------
// Kernel 1: h = relu(x @ W1 + b1)   [N,75]@[75,16] -> [N,16]
// LDS-staged: W1 transposed+padded to [16][96] f16, x tiles padded [16][96] f16.
// All WMMA fragment gathers are unconditional LDS loads (ds_load_b128 clauses).
// ---------------------------------------------------------------------------
__global__ void __launch_bounds__(128) k_lin1(
    const float* __restrict__ x, const float* __restrict__ W1,
    const float* __restrict__ b1, float* __restrict__ h, int n) {
  __shared__ _Float16 sW[16][96];                 // sW[col][k] = W1[k][col], 0-padded
  __shared__ _Float16 sX[TPB_TILES][16][96];      // per-tile rows, 0-padded K
  const int tid = threadIdx.x;
  const int lane = tid & 31, wave = tid >> 5;
  const int tile = blockIdx.x * TPB_TILES + wave;

  for (int idx = tid; idx < 16 * 96; idx += 128) {
    int c = idx / 96, k = idx - c * 96;
    sW[c][k] = (_Float16)(k < 75 ? W1[(size_t)k * 16 + c] : 0.f);
  }
  const bool tok = (tile * 16) < n;
  for (int r = 0; r < 16; ++r) {
    int row = tile * 16 + r;
    const bool rv = tok && (row < n);
    const float* xr = x + (size_t)row * 75;
    for (int k = lane; k < 96; k += 32)
      sX[wave][r][k] = (_Float16)((rv && k < 75) ? xr[k] : 0.f);
  }
  __syncthreads();
  if (!tok) return;

  const int m = lane & 15, half = lane >> 4;
  v8f acc = {};
#pragma unroll
  for (int chunk = 0; chunk < 3; ++chunk) {
    v16h a, b;
#pragma unroll
    for (int p = 0; p < 8; ++p) {
      // A 16x32 f16 layout: lanes0-15 V0-3:K0-7 V4-7:K16-23; lanes16-31: +8
      int ka = chunk * 32 + (p & 3) * 2 + (p >> 2) * 16 + half * 8;
      h2v av = *(const h2v*)&sX[wave][m][ka];
      a[2 * p] = av.x; a[2 * p + 1] = av.y;
      // B 32x16 f16 layout: lanes0-15 V0-3:K0-7 V4-7:K16-23; lanes16-31: +8
      int kb = chunk * 32 + (p & 3) * 2 + (p >> 2) * 16 + half * 8;
      h2v bv = *(const h2v*)&sW[m][kb];
      b[2 * p] = bv.x; b[2 * p + 1] = bv.y;
    }
    acc = __builtin_amdgcn_wmma_f32_16x16x32_f16(false, a, false, b, (short)0,
                                                 acc, false, false);
  }
  const float bias = b1[m];
  // C layout: row = r + 8*half, col = m. For fixed r, lanes 0-15 write one
  // contiguous 64B row segment -> already coalesced.
  float* hp = h + ((size_t)(tile * 16 + 8 * half)) * 16 + m;
  if (tile * 16 + 16 <= n) {                 // wave-uniform fast path: no guards
#pragma unroll
    for (int r = 0; r < 8; ++r) {
      float v = acc[r] + bias;
      hp[r * 16] = v > 0.f ? v : 0.f;
    }
  } else {
#pragma unroll
    for (int r = 0; r < 8; ++r) {
      int rr = tile * 16 + r + 8 * half;
      if (rr < n) {
        float v = acc[r] + bias;
        h[(size_t)rr * 16 + m] = v > 0.f ? v : 0.f;
      }
    }
  }
}

// ---------------------------------------------------------------------------
// Per node: xn = h/max(||h||,eps); self-loop exp seeded into s; zero h_next.
// ---------------------------------------------------------------------------
__global__ void k_norm_self(const float* __restrict__ h, float* __restrict__ xn,
                            float* __restrict__ s, float* __restrict__ eself,
                            float* __restrict__ hnext,
                            const float* beta_ptr, float beta_c, int n) {
  int i = blockIdx.x * blockDim.x + threadIdx.x;
  if (i >= n) return;
  float beta = beta_ptr ? beta_ptr[0] : beta_c;
  const float4* hr = (const float4*)(h + (size_t)i * 16);
  float4 v0 = hr[0], v1 = hr[1], v2 = hr[2], v3 = hr[3];
  float ss = v0.x * v0.x + v0.y * v0.y + v0.z * v0.z + v0.w * v0.w +
             v1.x * v1.x + v1.y * v1.y + v1.z * v1.z + v1.w * v1.w +
             v2.x * v2.x + v2.y * v2.y + v2.z * v2.z + v2.w * v2.w +
             v3.x * v3.x + v3.y * v3.y + v3.z * v3.z + v3.w * v3.w;
  float inv = 1.f / fmaxf(sqrtf(ss), EPS_N);
  float4* xr = (float4*)(xn + (size_t)i * 16);
  v0.x *= inv; v0.y *= inv; v0.z *= inv; v0.w *= inv;
  v1.x *= inv; v1.y *= inv; v1.z *= inv; v1.w *= inv;
  v2.x *= inv; v2.y *= inv; v2.z *= inv; v2.w *= inv;
  v3.x *= inv; v3.y *= inv; v3.z *= inv; v3.w *= inv;
  xr[0] = v0; xr[1] = v1; xr[2] = v2; xr[3] = v3;
  // self-loop logit = beta * dot(xn,xn); no segment-max needed (|logit|<=|beta|)
  float q = ss * inv * inv;
  float es = __expf(beta * q);
  s[i] = es; eself[i] = es;
  float4 z = {0.f, 0.f, 0.f, 0.f};
  float4* hb = (float4*)(hnext + (size_t)i * 16);
  hb[0] = z; hb[1] = z; hb[2] = z; hb[3] = z;
}

// ---------------------------------------------------------------------------
// Edge pass A: e = exp(beta * <xn[src],xn[dst]>); s[dst] += e (HW f32 atomic).
// ---------------------------------------------------------------------------
__global__ void k_edgeA(const int* __restrict__ src, const int* __restrict__ dst,
                        const float* __restrict__ xn, float* __restrict__ s,
                        float* __restrict__ elog,
                        const float* beta_ptr, float beta_c, int E) {
  int e = blockIdx.x * blockDim.x + threadIdx.x;
  if (e >= E) return;
  __builtin_prefetch(src + e + 4096, 0, 1);   // global_prefetch_b8
  __builtin_prefetch(dst + e + 4096, 0, 1);
  float beta = beta_ptr ? beta_ptr[0] : beta_c;
  int a = src[e], b = dst[e];
  const float4* pa = (const float4*)(xn + (size_t)a * 16);
  const float4* pb = (const float4*)(xn + (size_t)b * 16);
  float d = 0.f;
#pragma unroll
  for (int j = 0; j < 4; ++j) {
    float4 u = pa[j], v = pb[j];
    d += u.x * v.x + u.y * v.y + u.z * v.z + u.w * v.w;
  }
  float ee = __expf(beta * d);
  elog[e] = ee;
  unsafeAtomicAdd(&s[b], ee);                 // global_atomic_add_f32
}

// ---------------------------------------------------------------------------
// Edge pass B: hnext[dst] += (e/s[dst]) * h[src]
// ---------------------------------------------------------------------------
__global__ void k_edgeB(const int* __restrict__ src, const int* __restrict__ dst,
                        const float* __restrict__ h, const float* __restrict__ s,
                        const float* __restrict__ elog, float* __restrict__ hnext,
                        int E) {
  int e = blockIdx.x * blockDim.x + threadIdx.x;
  if (e >= E) return;
  __builtin_prefetch(elog + e + 4096, 0, 1);
  int a = src[e], b = dst[e];
  float alpha = elog[e] / s[b];
  const float4* ph = (const float4*)(h + (size_t)a * 16);
  float* out = hnext + (size_t)b * 16;
#pragma unroll
  for (int j = 0; j < 4; ++j) {
    float4 v = ph[j];
    unsafeAtomicAdd(out + 4 * j + 0, alpha * v.x);
    unsafeAtomicAdd(out + 4 * j + 1, alpha * v.y);
    unsafeAtomicAdd(out + 4 * j + 2, alpha * v.z);
    unsafeAtomicAdd(out + 4 * j + 3, alpha * v.w);
  }
}

// ---------------------------------------------------------------------------
// Self-loop contribution (after edge pass B; stream-ordered so no atomics).
// ---------------------------------------------------------------------------
__global__ void k_selfB(const float* __restrict__ h, const float* __restrict__ s,
                        const float* __restrict__ eself, float* __restrict__ hnext,
                        int n) {
  int i = blockIdx.x * blockDim.x + threadIdx.x;
  if (i >= n) return;
  float alpha = eself[i] / s[i];
  const float4* ph = (const float4*)(h + (size_t)i * 16);
  float4* po = (float4*)(hnext + (size_t)i * 16);
#pragma unroll
  for (int j = 0; j < 4; ++j) {
    float4 v = ph[j], o = po[j];
    o.x += alpha * v.x; o.y += alpha * v.y; o.z += alpha * v.z; o.w += alpha * v.w;
    po[j] = o;
  }
}

__global__ void k_inity(float* __restrict__ y, const float* __restrict__ bg, int g) {
  int i = blockIdx.x * blockDim.x + threadIdx.x;
  if (i < g) y[i] = bg[0];
}

// ---------------------------------------------------------------------------
// Fused tail: per 16-row tile, WMMA h@W2 (+b2) -> 16x64 in regs, dot rows with
// Wg, atomically pool into y[batch[row]].  h2 never hits memory.
// LDS-staged: W2 transposed+padded [64][32] f16, h tiles padded [16][32] f16.
// ---------------------------------------------------------------------------
__global__ void __launch_bounds__(128) k_lin2_pool(
    const float* __restrict__ h, const int* __restrict__ batch,
    const float* __restrict__ W2, const float* __restrict__ b2,
    const float* __restrict__ Wg, float* __restrict__ y, int n) {
  __shared__ _Float16 sW[64][32];                 // sW[col][k] = W2[k][col], 0-padded
  __shared__ _Float16 sH[TPB_TILES][16][32];      // per-tile rows, K padded 16->32
  const int tid = threadIdx.x;
  const int lane = tid & 31, wave = tid >> 5;
  const int tile = blockIdx.x * TPB_TILES + wave;

  for (int idx = tid; idx < 64 * 32; idx += 128) {
    int c = idx >> 5, k = idx & 31;
    sW[c][k] = (_Float16)(k < 16 ? W2[(size_t)k * 64 + c] : 0.f);
  }
  const bool tok = (tile * 16) < n;
  for (int r = 0; r < 16; ++r) {
    int row = tile * 16 + r;
    const bool rv = tok && (row < n);
    int k = lane;                                  // 32 lanes cover K=0..31
    sH[wave][r][k] = (_Float16)((rv && k < 16) ? h[(size_t)row * 16 + k] : 0.f);
  }
  __syncthreads();
  if (!tok) return;

  const int m = lane & 15, half = lane >> 4;
  v16h a;
#pragma unroll
  for (int p = 0; p < 8; ++p) {
    int ka = (p & 3) * 2 + (p >> 2) * 16 + half * 8;
    h2v av = *(const h2v*)&sH[wave][m][ka];
    a[2 * p] = av.x; a[2 * p + 1] = av.y;
  }
  float rp[8] = {0.f, 0.f, 0.f, 0.f, 0.f, 0.f, 0.f, 0.f};
#pragma unroll
  for (int t = 0; t < 4; ++t) {                    // 4 tiles of 16 output columns
    const int col = t * 16 + m;
    v16h b;
#pragma unroll
    for (int p = 0; p < 8; ++p) {
      int kb = (p & 3) * 2 + (p >> 2) * 16 + half * 8;
      h2v bv = *(const h2v*)&sW[col][kb];
      b[2 * p] = bv.x; b[2 * p + 1] = bv.y;
    }
    v8f c = {};
    c = __builtin_amdgcn_wmma_f32_16x16x32_f16(false, a, false, b, (short)0,
                                               c, false, false);
    float wg = Wg[col], bb = b2[col];
#pragma unroll
    for (int r = 0; r < 8; ++r) rp[r] += (c[r] + bb) * wg;
  }
  // Reduce across the 16 lanes of each half-group (offsets 1..8 preserve bit4).
#pragma unroll
  for (int off = 1; off < 16; off <<= 1) {
#pragma unroll
    for (int r = 0; r < 8; ++r) rp[r] += __shfl_xor(rp[r], off, 32);
  }
  if ((lane & 15) == 0) {
    if (tile * 16 + 16 <= n) {               // uniform fast path: no row guards
#pragma unroll
      for (int r = 0; r < 8; ++r) {
        int rr = tile * 16 + r + 8 * half;
        unsafeAtomicAdd(&y[batch[rr]], rp[r]);
      }
    } else {
#pragma unroll
      for (int r = 0; r < 8; ++r) {
        int rr = tile * 16 + r + 8 * half;
        if (rr < n) unsafeAtomicAdd(&y[batch[rr]], rp[r]);
      }
    }
  }
}

extern "C" void kernel_launch(void* const* d_in, const int* in_sizes, int n_in,
                              void* d_out, int out_size, void* d_ws, size_t ws_size,
                              hipStream_t stream) {
  (void)n_in; (void)ws_size;
  const float* x     = (const float*)d_in[0];
  const int*   ei    = (const int*)d_in[1];
  const int*   batch = (const int*)d_in[2];
  const float* W1    = (const float*)d_in[3];
  const float* b1    = (const float*)d_in[4];
  const float* beta2 = (const float*)d_in[5];
  const float* W2    = (const float*)d_in[6];
  const float* b2    = (const float*)d_in[7];
  const float* Wg    = (const float*)d_in[8];
  const float* bg    = (const float*)d_in[9];
  const int n = in_sizes[0] / 75;
  const int E = in_sizes[1] / 2;
  const int G = out_size;                    // [1024,1]
  const int* src = ei;
  const int* dst = ei + E;

  char* ws = (char*)d_ws;
  auto aln = [](size_t v) { return (v + 255) & ~(size_t)255; };
  const size_t nb16 = (size_t)n * 16 * sizeof(float);
  const size_t nb1  = (size_t)n * sizeof(float);
  float* h_a  = (float*)ws; ws += aln(nb16);
  float* h_b  = (float*)ws; ws += aln(nb16);
  float* xn   = (float*)ws; ws += aln(nb16);
  float* s    = (float*)ws; ws += aln(nb1);
  float* esf  = (float*)ws; ws += aln(nb1);
  float* elog = (float*)ws; ws += aln((size_t)E * sizeof(float));

  const int tiles = (n + 15) / 16;
  dim3 tb(128), tg((tiles + TPB_TILES - 1) / TPB_TILES);
  const int nt = 256;
  const int nbk = (n + nt - 1) / nt;
  const int ebk = (E + nt - 1) / nt;
  float* y = (float*)d_out;

  k_lin1<<<tg, tb, 0, stream>>>(x, W1, b1, h_a, n);
  // prop1 (beta fixed = 1)
  k_norm_self<<<nbk, nt, 0, stream>>>(h_a, xn, s, esf, h_b, nullptr, 1.0f, n);
  k_edgeA<<<ebk, nt, 0, stream>>>(src, dst, xn, s, elog, nullptr, 1.0f, E);
  k_edgeB<<<ebk, nt, 0, stream>>>(src, dst, h_a, s, elog, h_b, E);
  k_selfB<<<nbk, nt, 0, stream>>>(h_a, s, esf, h_b, n);
  // prop2 (learnable beta2, read on device)
  k_norm_self<<<nbk, nt, 0, stream>>>(h_b, xn, s, esf, h_a, beta2, 1.0f, n);
  k_edgeA<<<ebk, nt, 0, stream>>>(src, dst, xn, s, elog, beta2, 1.0f, E);
  k_edgeB<<<ebk, nt, 0, stream>>>(src, dst, h_b, s, elog, h_a, E);
  k_selfB<<<nbk, nt, 0, stream>>>(h_b, s, esf, h_a, n);
  // fused tail
  k_inity<<<(G + 255) / 256, 256, 0, stream>>>(y, bg, G);
  k_lin2_pool<<<tg, tb, 0, stream>>>(h_a, batch, W2, b2, Wg, y, n);
}
